// HGConv_27169963114587
// MI455X (gfx1250) — matrix operand order repaired and checked by reading, not compiled, not run on hardware
//
#include <hip/hip_runtime.h>
#include <hip/hip_bf16.h>

// Problem dimensions (fixed by the reference).
#define Bsz 8
#define Mm  4096
#define Ee  1024
#define Dd  256
#define Cc  16
#define ET  4   // e-tiles per wave in the dominant GEMM
#define MT  4   // m-tiles per wave in the scores GEMM

typedef unsigned short u16;
typedef __attribute__((ext_vector_type(8)))  u16    v8u;
typedef __attribute__((ext_vector_type(16))) u16    v16u;
typedef __attribute__((ext_vector_type(8)))  float  v8f;
typedef __attribute__((ext_vector_type(16))) __bf16 v16bf;

static __device__ __forceinline__ v16bf as_bf(v16u u) {
  union U { v16u u; v16bf b; } c; c.u = u; return c.b;
}
// fp32 -> bf16 round-to-nearest-even
static __device__ __forceinline__ u16 f2bf(float x) {
  union { float f; unsigned u; } c; c.f = x;
  return (u16)((c.u + 0x7FFFu + ((c.u >> 16) & 1u)) >> 16);
}

// ---- WMMA fragment loaders (wave32; contiguous k order, same for A and B) ----
// A: 16x32 bf16, source row-major (row = M index, K contiguous).
//   lane (hi*16+m): slots 0..15 = K kk+16*hi .. kk+16*hi+15
static __device__ __forceinline__ v16bf load_A(const u16* __restrict__ row, int kk, int hi) {
  return as_bf(*(const v16u*)(row + kk + hi * 16));
}
// B: 32x16 bf16 where B(k,n) = srcT[n*ld + k] (K contiguous per lane n).
//   lane (hi*16+n): slots 0..15 = K kk+16*hi .. kk+16*hi+15
static __device__ __forceinline__ v16bf load_B(const u16* __restrict__ row, int kk, int hi) {
  return as_bf(*(const v16u*)(row + kk + hi * 16));
}
static __device__ __forceinline__ v8f wmma_bf16(v16bf a, v16bf b, v8f c) {
  return __builtin_amdgcn_wmma_f32_16x16x32_bf16(false, a, false, b, (short)0, c,
                                                 false, false);
}

// ---------------- prep kernels ----------------
__global__ void k_f32_to_bf16(const float* __restrict__ src, u16* __restrict__ dst, int n) {
  int i = blockIdx.x * blockDim.x + threadIdx.x;
  if (i < n) dst[i] = f2bf(src[i]);
}

// batched transpose: src (R,C) fp32 row-major -> dst (C,R) bf16 row-major
__global__ void k_transpose_f32_bf16(const float* __restrict__ src, u16* __restrict__ dst,
                                     int R, int C) {
  __shared__ float tile[32][33];
  int b = blockIdx.z;
  int c0 = blockIdx.x * 32, r0 = blockIdx.y * 32;
  const float* s = src + (size_t)b * R * C;
  u16* d = dst + (size_t)b * R * C;
#pragma unroll
  for (int i = 0; i < 4; ++i) {
    int r = r0 + threadIdx.y + i * 8;
    tile[threadIdx.y + i * 8][threadIdx.x] = s[(size_t)r * C + c0 + threadIdx.x];
  }
  __syncthreads();
#pragma unroll
  for (int i = 0; i < 4; ++i) {
    int c = c0 + threadIdx.y + i * 8;
    d[(size_t)c * R + r0 + threadIdx.x] = f2bf(tile[threadIdx.x][threadIdx.y + i * 8]);
  }
}

// ---------------- WMMA GEMM kernels ----------------
// scoresT[b][d'][m] = sum_d Wa[d'][d] * node[m][d]   (output directly transposed)
// One wave: MT m-tiles x 1 d'-tile; A (Wa rows) shared; manual double-buffer.
__global__ void k_gemm_scoresT(const u16* __restrict__ Wa_bf, const u16* __restrict__ node_bf,
                               u16* __restrict__ scoresT) {
  int lane = threadIdx.x & 31, lo = lane & 15, hi = lane >> 4;
  int mg = blockIdx.x, dt = blockIdx.y, b = blockIdx.z;
  const u16* Arow = Wa_bf + (size_t)(dt * 16 + lo) * Dd;
  const u16* Brow[MT];
#pragma unroll
  for (int i = 0; i < MT; ++i)
    Brow[i] = node_bf + ((size_t)b * Mm + mg * (16 * MT) + i * 16 + lo) * Dd;
  v8f acc[MT];
#pragma unroll
  for (int i = 0; i < MT; ++i) acc[i] = (v8f){};

  v16bf aC = load_A(Arow, 0, hi);
  v16bf bC[MT];
#pragma unroll
  for (int i = 0; i < MT; ++i) bC[i] = load_B(Brow[i], 0, hi);
  for (int kk = 0; kk < Dd - 32; kk += 32) {
    v16bf aN = load_A(Arow, kk + 32, hi);
    v16bf bN[MT];
#pragma unroll
    for (int i = 0; i < MT; ++i) bN[i] = load_B(Brow[i], kk + 32, hi);
#pragma unroll
    for (int i = 0; i < MT; ++i) acc[i] = wmma_bf16(aC, bC[i], acc[i]);
    aC = aN;
#pragma unroll
    for (int i = 0; i < MT; ++i) bC[i] = bN[i];
  }
#pragma unroll
  for (int i = 0; i < MT; ++i) acc[i] = wmma_bf16(aC, bC[i], acc[i]);

#pragma unroll
  for (int i = 0; i < MT; ++i) {
    u16* out = scoresT + ((size_t)b * Dd + dt * 16 + hi * 8) * Mm +
               mg * (16 * MT) + i * 16 + lo;
#pragma unroll
    for (int r = 0; r < 8; ++r) out[(size_t)r * Mm] = f2bf(acc[i][r]);
  }
}

// Fused dominant einsums (B fragments shared across ET e-tiles):
//   edge_scores[e][d] = sum_m incT[e][m]*scoresT[d][m]
//   agg[e][d]         = sum_m incT[e][m]*nodeT[d][m]
// stored in (b,d,e) layout so the e-softmax is contiguous. Double-buffered.
__global__ void k_gemm_edge(const u16* __restrict__ incT, const u16* __restrict__ scoresT,
                            const u16* __restrict__ nodeT,
                            float* __restrict__ sc_dE, float* __restrict__ ag_dE) {
  int lane = threadIdx.x & 31, lo = lane & 15, hi = lane >> 4;
  int eg = blockIdx.x, dt = blockIdx.y, b = blockIdx.z;
  const u16* Arow[ET];
#pragma unroll
  for (int i = 0; i < ET; ++i)
    Arow[i] = incT + ((size_t)b * Ee + eg * (16 * ET) + i * 16 + lo) * Mm;
  const u16* BsRow = scoresT + ((size_t)b * Dd + dt * 16 + lo) * Mm;
  const u16* BaRow = nodeT   + ((size_t)b * Dd + dt * 16 + lo) * Mm;
  v8f accS[ET], accA[ET];
#pragma unroll
  for (int i = 0; i < ET; ++i) { accS[i] = (v8f){}; accA[i] = (v8f){}; }

  v16bf bsC = load_B(BsRow, 0, hi);
  v16bf baC = load_B(BaRow, 0, hi);
  v16bf aC[ET];
#pragma unroll
  for (int i = 0; i < ET; ++i) aC[i] = load_A(Arow[i], 0, hi);

  for (int kk = 0; kk < Mm - 32; kk += 32) {
    // issue next-step loads before consuming current fragments
    v16bf bsN = load_B(BsRow, kk + 32, hi);
    v16bf baN = load_B(BaRow, kk + 32, hi);
    v16bf aN[ET];
#pragma unroll
    for (int i = 0; i < ET; ++i) aN[i] = load_A(Arow[i], kk + 32, hi);
#pragma unroll
    for (int i = 0; i < ET; ++i) {
      accS[i] = wmma_bf16(aC[i], bsC, accS[i]);
      accA[i] = wmma_bf16(aC[i], baC, accA[i]);
    }
    bsC = bsN; baC = baN;
#pragma unroll
    for (int i = 0; i < ET; ++i) aC[i] = aN[i];
  }
#pragma unroll
  for (int i = 0; i < ET; ++i) {
    accS[i] = wmma_bf16(aC[i], bsC, accS[i]);
    accA[i] = wmma_bf16(aC[i], baC, accA[i]);
  }

#pragma unroll
  for (int i = 0; i < ET; ++i) {
    size_t base = ((size_t)b * Dd + dt * 16 + lo) * Ee +
                  eg * (16 * ET) + i * 16 + hi * 8;
    *(float4*)(sc_dE + base)     = make_float4(accS[i][0], accS[i][1], accS[i][2], accS[i][3]);
    *(float4*)(sc_dE + base + 4) = make_float4(accS[i][4], accS[i][5], accS[i][6], accS[i][7]);
    *(float4*)(ag_dE + base)     = make_float4(accA[i][0], accA[i][1], accA[i][2], accA[i][3]);
    *(float4*)(ag_dE + base + 4) = make_float4(accA[i][4], accA[i][5], accA[i][6], accA[i][7]);
  }
}

// softmax over e (contiguous row of 1024) + multiply by agg; emit P bf16 in (b,e,d)
__global__ void k_softmax_scale(const float* __restrict__ sc_dE, const float* __restrict__ ag_dE,
                                u16* __restrict__ Pbf) {
  __shared__ float red[256];
  int row = blockIdx.x;           // = b*Dd + d
  int b = row / Dd, d = row % Dd;
  int t = threadIdx.x;
  float4 v = ((const float4*)(sc_dE + (size_t)row * Ee))[t];
  float m = fmaxf(fmaxf(v.x, v.y), fmaxf(v.z, v.w));
  red[t] = m; __syncthreads();
  for (int s = 128; s > 0; s >>= 1) { if (t < s) red[t] = fmaxf(red[t], red[t + s]); __syncthreads(); }
  m = red[0]; __syncthreads();
  float e0 = __expf(v.x - m), e1 = __expf(v.y - m), e2 = __expf(v.z - m), e3 = __expf(v.w - m);
  red[t] = e0 + e1 + e2 + e3; __syncthreads();
  for (int s = 128; s > 0; s >>= 1) { if (t < s) red[t] += red[t + s]; __syncthreads(); }
  float inv = 1.0f / red[0];
  float4 g = ((const float4*)(ag_dE + (size_t)row * Ee))[t];
  size_t base = ((size_t)b * Ee + t * 4) * Dd + d;
  Pbf[base         ] = f2bf(g.x * e0 * inv);
  Pbf[base + Dd    ] = f2bf(g.y * e1 * inv);
  Pbf[base + 2 * Dd] = f2bf(g.z * e2 * inv);
  Pbf[base + 3 * Dd] = f2bf(g.w * e3 * inv);
}

// ef[b][e][d'] = alpha*edge_feats + (1-alpha)*sum_d P[e][d]*Wp[d'][d]
__global__ void k_gemm_ef(const u16* __restrict__ Pbf, const u16* __restrict__ Wp_bf,
                          const float* __restrict__ edge_feats, const float* __restrict__ alpha_p,
                          float* __restrict__ ef_ws) {
  int lane = threadIdx.x & 31, lo = lane & 15, hi = lane >> 4;
  int et = blockIdx.x, dt = blockIdx.y, b = blockIdx.z;
  const u16* Arow = Pbf + ((size_t)b * Ee + et * 16 + lo) * Dd;
  const u16* Brow = Wp_bf + (size_t)(dt * 16 + lo) * Dd;
  v8f acc = {};
  v16bf aC = load_A(Arow, 0, hi);
  v16bf bC = load_B(Brow, 0, hi);
  for (int kk = 0; kk < Dd - 32; kk += 32) {
    v16bf aN = load_A(Arow, kk + 32, hi);
    v16bf bN = load_B(Brow, kk + 32, hi);
    acc = wmma_bf16(aC, bC, acc);
    aC = aN; bC = bN;
  }
  acc = wmma_bf16(aC, bC, acc);
  float alpha = *alpha_p;
  float beta = 1.0f - alpha;
#pragma unroll
  for (int r = 0; r < 8; ++r) {
    size_t idx = ((size_t)b * Ee + et * 16 + hi * 8 + r) * Dd + dt * 16 + lo;
    ef_ws[idx] = alpha * edge_feats[idx] + beta * acc[r];
  }
}

// ---------------- tail (bandwidth-trivial) ----------------
// att[b][e] = dot(ef[b][e][:], w_att)
__global__ void k_att_logits(const float* __restrict__ ef_ws, const float* __restrict__ watt,
                             float* __restrict__ att) {
  int w = threadIdx.x >> 5, lane = threadIdx.x & 31;
  int idx = blockIdx.x * 8 + w;  // 0 .. B*E-1
  const float4* p = (const float4*)(ef_ws + (size_t)idx * Dd + lane * 8);
  const float4* q = (const float4*)(watt + lane * 8);
  float4 a0 = p[0], a1 = p[1], b0 = q[0], b1 = q[1];
  float s = a0.x * b0.x + a0.y * b0.y + a0.z * b0.z + a0.w * b0.w +
            a1.x * b1.x + a1.y * b1.y + a1.z * b1.z + a1.w * b1.w;
  for (int off = 16; off > 0; off >>= 1) s += __shfl_xor(s, off);
  if (lane == 0) att[idx] = s;
}

// per-batch softmax over e + pooled[b][d] = sum_e ef[b][e][d]*a[e]
__global__ void k_softmax_pool(const float* __restrict__ att, const float* __restrict__ ef_ws,
                               float* __restrict__ pooled) {
  __shared__ float a_s[Ee];
  __shared__ float red[256];
  int b = blockIdx.x, t = threadIdx.x;
  float4 v = ((const float4*)(att + (size_t)b * Ee))[t];
  float m = fmaxf(fmaxf(v.x, v.y), fmaxf(v.z, v.w));
  red[t] = m; __syncthreads();
  for (int s = 128; s > 0; s >>= 1) { if (t < s) red[t] = fmaxf(red[t], red[t + s]); __syncthreads(); }
  m = red[0]; __syncthreads();
  float e0 = __expf(v.x - m), e1 = __expf(v.y - m), e2 = __expf(v.z - m), e3 = __expf(v.w - m);
  red[t] = e0 + e1 + e2 + e3; __syncthreads();
  for (int s = 128; s > 0; s >>= 1) { if (t < s) red[t] += red[t + s]; __syncthreads(); }
  float inv = 1.0f / red[0];
  a_s[t * 4 + 0] = e0 * inv; a_s[t * 4 + 1] = e1 * inv;
  a_s[t * 4 + 2] = e2 * inv; a_s[t * 4 + 3] = e3 * inv;
  __syncthreads();
  float acc = 0.f;
  const float* basep = ef_ws + (size_t)b * Ee * Dd + t;   // d = t, coalesced
  for (int e = 0; e < Ee; ++e) acc += basep[(size_t)e * Dd] * a_s[e];
  pooled[(size_t)b * Dd + t] = acc;
}

// out[b][d'] = dot(pooled[b], W[d']) + bias[d']
__global__ void k_proj(const float* __restrict__ pooled, const float* __restrict__ W,
                       const float* __restrict__ bias, float* __restrict__ out) {
  int b = blockIdx.x, d2 = threadIdx.x;
  const float4* x = (const float4*)(pooled + (size_t)b * Dd);
  const float4* w = (const float4*)(W + (size_t)d2 * Dd);
  float acc = 0.f;
  for (int i = 0; i < Dd / 4; ++i) {
    float4 a = x[i], c = w[i];
    acc += a.x * c.x + a.y * c.y + a.z * c.z + a.w * c.w;
  }
  out[(size_t)b * Dd + d2] = acc + bias[d2];
}

// logits[b][c] = dot(out[b], fc_w[c]) + fc_b[c]
__global__ void k_fc(const float* __restrict__ outp, const float* __restrict__ W,
                     const float* __restrict__ bias, float* __restrict__ logits) {
  int t = threadIdx.x;  // 0..127 => b = t/16, c = t%16
  int b = t >> 4, c = t & 15;
  const float4* x = (const float4*)(outp + (size_t)b * Dd);
  const float4* w = (const float4*)(W + (size_t)c * Dd);
  float acc = 0.f;
  for (int i = 0; i < Dd / 4; ++i) {
    float4 a = x[i], d = w[i];
    acc += a.x * d.x + a.y * d.y + a.z * d.z + a.w * d.w;
  }
  logits[t] = acc + bias[c];
}

extern "C" void kernel_launch(void* const* d_in, const int* in_sizes, int n_in,
                              void* d_out, int out_size, void* d_ws, size_t ws_size,
                              hipStream_t stream) {
  (void)in_sizes; (void)n_in; (void)out_size; (void)ws_size;
  const float* node_feats = (const float*)d_in[0];
  const float* edge_feats = (const float*)d_in[1];
  const float* inc_mat    = (const float*)d_in[2];
  const float* Wa         = (const float*)d_in[3];
  const float* Wp         = (const float*)d_in[4];
  const float* alpha      = (const float*)d_in[5];
  const float* ec_att_w   = (const float*)d_in[6];
  const float* ec_proj_w  = (const float*)d_in[7];
  const float* ec_proj_b  = (const float*)d_in[8];
  const float* fc_w       = (const float*)d_in[9];
  const float* fc_b       = (const float*)d_in[10];
  float* logits = (float*)d_out;

  size_t off = 0;
  auto carve = [&](size_t bytes) -> char* {
    char* p = (char*)d_ws + off;
    off += (bytes + 255) & ~(size_t)255;
    return p;
  };
  u16*   Wa_bf   = (u16*)  carve((size_t)Dd * Dd * 2);
  u16*   Wp_bf   = (u16*)  carve((size_t)Dd * Dd * 2);
  u16*   node_bf = (u16*)  carve((size_t)Bsz * Mm * Dd * 2);
  u16*   nodeT   = (u16*)  carve((size_t)Bsz * Dd * Mm * 2);
  u16*   incT    = (u16*)  carve((size_t)Bsz * Ee * Mm * 2);
  u16*   scoresT = (u16*)  carve((size_t)Bsz * Dd * Mm * 2);
  float* sc_dE   = (float*)carve((size_t)Bsz * Dd * Ee * 4);
  float* ag_dE   = (float*)carve((size_t)Bsz * Dd * Ee * 4);
  u16*   Pbf     = (u16*)  carve((size_t)Bsz * Ee * Dd * 2);
  float* ef_ws   = (float*)carve((size_t)Bsz * Ee * Dd * 4);
  float* att     = (float*)carve((size_t)Bsz * Ee * 4);
  float* pooled  = (float*)carve((size_t)Bsz * Dd * 4);
  float* outp    = (float*)carve((size_t)Bsz * Dd * 4);

  // --- prep: bf16 conversions + transposes ---
  int nW = Dd * Dd;
  k_f32_to_bf16<<<(nW + 255) / 256, 256, 0, stream>>>(Wa, Wa_bf, nW);
  k_f32_to_bf16<<<(nW + 255) / 256, 256, 0, stream>>>(Wp, Wp_bf, nW);
  int nN = Bsz * Mm * Dd;
  k_f32_to_bf16<<<(nN + 255) / 256, 256, 0, stream>>>(node_feats, node_bf, nN);
  k_transpose_f32_bf16<<<dim3(Ee / 32, Mm / 32, Bsz), dim3(32, 8), 0, stream>>>(
      inc_mat, incT, Mm, Ee);
  k_transpose_f32_bf16<<<dim3(Dd / 32, Mm / 32, Bsz), dim3(32, 8), 0, stream>>>(
      node_feats, nodeT, Mm, Dd);

  // --- node attention GEMM (output transposed) ---
  k_gemm_scoresT<<<dim3(Mm / (16 * MT), Dd / 16, Bsz), 32, 0, stream>>>(Wa_bf, node_bf,
                                                                        scoresT);

  // --- fused dominant einsums via WMMA ---
  k_gemm_edge<<<dim3(Ee / (16 * ET), Dd / 16, Bsz), 32, 0, stream>>>(incT, scoresT, nodeT,
                                                                     sc_dE, ag_dE);

  // --- e-softmax * agg -> P (bf16) ---
  k_softmax_scale<<<Bsz * Dd, 256, 0, stream>>>(sc_dE, ag_dE, Pbf);

  // --- projection GEMM + alpha blend ---
  k_gemm_ef<<<dim3(Ee / 16, Dd / 16, Bsz), 32, 0, stream>>>(Pbf, Wp_bf, edge_feats,
                                                            alpha, ef_ws);

  // --- edge aggregation + classifier tail ---
  k_att_logits<<<Bsz * Ee / 8, 256, 0, stream>>>(ef_ws, ec_att_w, att);
  k_softmax_pool<<<Bsz, 256, 0, stream>>>(att, ef_ws, pooled);
  k_proj<<<Bsz, Dd, 0, stream>>>(pooled, ec_proj_w, ec_proj_b, outp);
  k_fc<<<1, Bsz * Cc, 0, stream>>>(outp, fc_w, fc_b, logits);
}
